// SeqDecoderLSTM_51367808860617
// MI455X (gfx1250) — matrix operand order repaired and checked by reading, not compiled, not run on hardware
//
#include <hip/hip_runtime.h>
#include <hip/hip_bf16.h>

// ---------------------------------------------------------------------------
// SeqDecoderLSTM on MI455X (gfx1250, wave32, WMMA).
// T=64 sequential steps; ~310 GFLOP total -> latency bound. Weights converted
// once to bf16 (~5MB, L2-resident under 192MB L2). All matmuls use
// v_wmma_f32_16x16x32_bf16 with fp32 accumulation.
// ---------------------------------------------------------------------------

typedef __attribute__((ext_vector_type(16))) __bf16 v16bf;
typedef __attribute__((ext_vector_type(8)))  float  v8f;

union Frag {          // one WMMA 16-bit operand: 16 bf16 per lane = 32 bytes
  v16bf v;
  uint4 q[2];
};

static constexpr int T = 64, B = 1024, H = 512, V = 512;
static constexpr int FOURH = 4 * H;   // 2048 gate columns
static constexpr int K2    = 2 * H;   // 1024 = width of xh = [x | h]

// workspace layout (bytes), every region 256B-aligned by construction
static constexpr size_t OFF_WCAT = 0;                          // [4H,2H] bf16
static constexpr size_t SZ_WCAT  = (size_t)FOURH * K2 * 2;     // 4 MB
static constexpr size_t OFF_WOUT = OFF_WCAT + SZ_WCAT;         // [V,H] bf16
static constexpr size_t SZ_WOUT  = (size_t)V * H * 2;
static constexpr size_t OFF_EMB  = OFF_WOUT + SZ_WOUT;         // [V,H] bf16
static constexpr size_t SZ_EMB   = (size_t)V * H * 2;
static constexpr size_t OFF_BIAS = OFF_EMB + SZ_EMB;           // [4H] f32
static constexpr size_t SZ_BIAS  = (size_t)FOURH * 4;
static constexpr size_t OFF_XH0  = OFF_BIAS + SZ_BIAS;         // [B,2H] bf16
static constexpr size_t SZ_XH    = (size_t)B * K2 * 2;
static constexpr size_t OFF_XH1  = OFF_XH0 + SZ_XH;
static constexpr size_t OFF_C    = OFF_XH1 + SZ_XH;            // [B,H] f32

__device__ __forceinline__ unsigned short f2bf(float x) {
  union { float f; unsigned u; } v; v.f = x;
  unsigned r = v.u + 0x7FFFu + ((v.u >> 16) & 1u);   // round-to-nearest-even
  return (unsigned short)(r >> 16);
}

// ---------------------------------------------------------------------------
// One-time prep kernels
// ---------------------------------------------------------------------------
__global__ void k_prep_wcat(const float* __restrict__ Wih,
                            const float* __restrict__ Whh,
                            unsigned short* __restrict__ Wcat) {
  int idx = blockIdx.x * 256 + threadIdx.x;        // [0, 4H*2H)
  int r = idx >> 10, c = idx & 1023;
  float v = (c < H) ? Wih[r * H + c] : Whh[r * H + (c - H)];
  Wcat[idx] = f2bf(v);
}

__global__ void k_prep_misc(const float* __restrict__ Wout,
                            const float* __restrict__ Emb,
                            const float* __restrict__ bih,
                            const float* __restrict__ bhh,
                            unsigned short* __restrict__ WoutB,
                            unsigned short* __restrict__ EmbB,
                            float* __restrict__ bias) {
  int idx = blockIdx.x * 256 + threadIdx.x;
  if (idx < V * H)               WoutB[idx] = f2bf(Wout[idx]);
  else if (idx < 2 * V * H)      EmbB[idx - V * H] = f2bf(Emb[idx - V * H]);
  else if (idx < 2 * V * H + FOURH) {
    int i = idx - 2 * V * H;
    bias[i] = bih[i] + bhh[i];
  }
}

__global__ void k_init_state(const float* __restrict__ encH,
                             const float* __restrict__ encC,
                             const float* __restrict__ Emb,
                             unsigned short* __restrict__ xh0,
                             float* __restrict__ Cst) {
  int idx = blockIdx.x * 256 + threadIdx.x;        // [0, B*H)
  int b = idx >> 9, j = idx & 511;
  Cst[idx] = encC[idx];
  xh0[(size_t)b * K2 + H + j] = f2bf(encH[idx]);   // h0
  xh0[(size_t)b * K2 + j]     = f2bf(Emb[j]);      // x0 = embedding[SOS=0]
}

// ---------------------------------------------------------------------------
// Kernel A: gates = xh @ Wcat^T + bias (K=1024, single fused GEMM), then LSTM
// cell. Block tile: 64 batch rows x 32 hidden cols (= 128 gate cols, 4 gates
// grouped so i/f/g/o of a hidden unit are local). 8 waves: 4 M-groups x 2
// N-groups; each wave -> one 16-row A frag, four 16-col B frags per K step.
// ---------------------------------------------------------------------------
__global__ __launch_bounds__(256)
void k_lstm_step(const unsigned short* __restrict__ xh_in,
                 unsigned short* __restrict__ xh_out,
                 const unsigned short* __restrict__ Wcat,
                 const float* __restrict__ bias,
                 float* __restrict__ Cst) {
  constexpr int MT = 64;
  constexpr int AS = 40;                           // padded LDS row stride
  __shared__ __align__(16) unsigned short As[MT * AS];    //  5 KB
  __shared__ __align__(16) unsigned short Bs[128 * AS];   // 10 KB
  __shared__ float Gs[MT * 132];                          // 33 KB gates

  const int tid  = threadIdx.x;
  const int lane = tid & 31, w = tid >> 5;
  const int m0g  = blockIdx.x * MT;                // batch row base
  const int j0   = blockIdx.y * 32;                // hidden col base
  const int wm   = (w & 3) * 16;                   // wave M offset
  const int wn   = (w >> 2) * 64;                  // wave N offset (0/64)

  v8f acc[4] = {};                                 // 4 N-subtiles of 16

  const int ar = tid >> 2, ac = (tid & 3) * 8;     // A loader: 64r x 4x8
  const int br = tid >> 1, bc = (tid & 1) * 16;    // B loader: 128r x 2x16
  const int bw_row = (br >> 5) * H + j0 + (br & 31); // gate*H + hidden col

  for (int k0 = 0; k0 < K2; k0 += 32) {
    *(uint4*)(As + ar * AS + ac) =
        *(const uint4*)(xh_in + (size_t)(m0g + ar) * K2 + k0 + ac);
    const uint4* bsrc = (const uint4*)(Wcat + (size_t)bw_row * K2 + k0 + bc);
    *(uint4*)(Bs + br * AS + bc)     = bsrc[0];
    *(uint4*)(Bs + br * AS + bc + 8) = bsrc[1];
    if (k0 + 32 < K2)                               // global_prefetch_b8
      __builtin_prefetch(Wcat + (size_t)bw_row * K2 + k0 + 32 + bc, 0, 1);
    __syncthreads();

    // A frag: 16x32 bf16, 16 contiguous K per lane (lanes 0-15: K 0..15, 16-31: K 16..31)
    Frag a;
    {
      const unsigned short* p =
          As + (wm + (lane & 15)) * AS + ((lane < 16) ? 0 : 16);
      a.q[0] = *(const uint4*)p;
      a.q[1] = *(const uint4*)(p + 8);
    }
#pragma unroll
    for (int nt = 0; nt < 4; ++nt) {
      // B frag: 32x16, lane = N, 16 contiguous K values per lane
      Frag bf;
      const unsigned short* p =
          Bs + (wn + nt * 16 + (lane & 15)) * AS + ((lane < 16) ? 0 : 16);
      bf.q[0] = *(const uint4*)p;
      bf.q[1] = *(const uint4*)(p + 8);
      acc[nt] = __builtin_amdgcn_wmma_f32_16x16x32_bf16(
          false, a.v, false, bf.v, (short)0, acc[nt], false, false);
    }
    __syncthreads();
  }

  // spill gates (+bias) to LDS using the ISA C/D layout (vgpr r -> row r/r+8)
#pragma unroll
  for (int nt = 0; nt < 4; ++nt) {
    int colL = wn + nt * 16 + (lane & 15);         // 0..127
    float bv = bias[(colL >> 5) * H + j0 + (colL & 31)];
#pragma unroll
    for (int r = 0; r < 8; ++r) {
      int rowL = wm + r + ((lane >> 4) << 3);
      Gs[rowL * 132 + colL] = acc[nt][r] + bv;
    }
  }
  __syncthreads();

  // fused LSTM cell: c_new = sig(f)*c + sig(i)*tanh(g); h_new = sig(o)*tanh(c)
  for (int e = tid; e < MT * 32; e += 256) {
    int rb = e >> 5, hc = e & 31;
    float gi = Gs[rb * 132 + hc];
    float gf = Gs[rb * 132 + 32 + hc];
    float gg = Gs[rb * 132 + 64 + hc];
    float go = Gs[rb * 132 + 96 + hc];
    size_t cix = (size_t)(m0g + rb) * H + (j0 + hc);
    float si = 1.f / (1.f + __expf(-gi));
    float sf = 1.f / (1.f + __expf(-gf));
    float so = 1.f / (1.f + __expf(-go));
    float cn = sf * Cst[cix] + si * tanhf(gg);
    float hn = so * tanhf(cn);
    Cst[cix] = cn;
    xh_out[(size_t)(m0g + rb) * K2 + H + (j0 + hc)] = f2bf(hn);
  }
}

// ---------------------------------------------------------------------------
// Kernel B: logits = h_new @ Wout^T + b_out, argmax over V (first-max ties),
// embedding gather -> x part of next xh. One block = 16 batch rows x all 512
// vocab cols (argmax stays local). Frags loaded straight from L2-resident
// bf16 weights. 8 waves x 64 cols.
// ---------------------------------------------------------------------------
__global__ __launch_bounds__(256)
void k_logits_argmax(unsigned short* __restrict__ xh,   // reads h, writes x
                     const unsigned short* __restrict__ WoutB,
                     const float* __restrict__ b_out,
                     const unsigned short* __restrict__ EmbB,
                     float* __restrict__ out_t) {
  constexpr int LST = 516;
  __shared__ float Ls[16 * LST];                   // 33 KB logits
  __shared__ float rMax[16][16];
  __shared__ int   rIdx[16][16];
  __shared__ int   best[16];

  const int tid = threadIdx.x, lane = tid & 31, w = tid >> 5;
  const int m0 = blockIdx.x * 16;
  const int wn = w * 64;

  v8f acc[4] = {};
  const size_t arow  = (size_t)(m0 + (lane & 15)) * K2 + H;  // h part
  const int    akoff = (lane < 16) ? 0 : 16;
  const int    bkoff = (lane < 16) ? 0 : 16;

  for (int k0 = 0; k0 < H; k0 += 32) {
    Frag a;
    const unsigned short* pa = xh + arow + k0 + akoff;
    a.q[0] = *(const uint4*)pa;
    a.q[1] = *(const uint4*)(pa + 8);
#pragma unroll
    for (int nt = 0; nt < 4; ++nt) {
      Frag bf;
      const unsigned short* pb =
          WoutB + (size_t)(wn + nt * 16 + (lane & 15)) * H + k0 + bkoff;
      bf.q[0] = *(const uint4*)pb;
      bf.q[1] = *(const uint4*)(pb + 8);
      acc[nt] = __builtin_amdgcn_wmma_f32_16x16x32_bf16(
          false, a.v, false, bf.v, (short)0, acc[nt], false, false);
    }
  }

#pragma unroll
  for (int nt = 0; nt < 4; ++nt) {
    int col = wn + nt * 16 + (lane & 15);
    float bv = b_out[col];
#pragma unroll
    for (int r = 0; r < 8; ++r) {
      int row = r + ((lane >> 4) << 3);
      float vv = acc[nt][r] + bv;
      out_t[(size_t)(m0 + row) * V + col] = vv;
      Ls[row * LST + col] = vv;
    }
  }
  __syncthreads();

  {  // per-row argmax, 16 strided partitions per row, first-max tie rule
    int row = tid >> 4, sub = tid & 15;
    float bv = -3.4e38f; int bi = 0;
    for (int c = sub; c < V; c += 16) {
      float vv = Ls[row * LST + c];
      if (vv > bv) { bv = vv; bi = c; }
    }
    rMax[row][sub] = bv; rIdx[row][sub] = bi;
  }
  __syncthreads();
  if ((tid & 15) == 0) {
    int row = tid >> 4;
    float bv = rMax[row][0]; int bi = rIdx[row][0];
    for (int s = 1; s < 16; ++s) {
      float vv = rMax[row][s]; int ii = rIdx[row][s];
      if (vv > bv || (vv == bv && ii < bi)) { bv = vv; bi = ii; }
    }
    best[row] = bi;
  }
  __syncthreads();
  {  // gather embedding[argmax] (bf16) -> x part; 64B per thread
    int row = tid >> 4, sub = tid & 15;
    const uint4* s4 = (const uint4*)(EmbB + (size_t)best[row] * H + sub * 32);
    uint4* d4 = (uint4*)(xh + (size_t)(m0 + row) * K2 + sub * 32);
    d4[0] = s4[0]; d4[1] = s4[1]; d4[2] = s4[2]; d4[3] = s4[3];
  }
}

// ---------------------------------------------------------------------------
extern "C" void kernel_launch(void* const* d_in, const int* in_sizes, int n_in,
                              void* d_out, int out_size, void* d_ws, size_t ws_size,
                              hipStream_t stream) {
  // setup_inputs order: 0 target_var(i64, unused) 1 target_max_len(unused)
  // 2 encoder_hidden 3 encoder_cell 4 embedding 5 W_ih 6 W_hh 7 b_ih 8 b_hh
  // 9 W_out 10 b_out
  const float* encH = (const float*)d_in[2];
  const float* encC = (const float*)d_in[3];
  const float* Emb  = (const float*)d_in[4];
  const float* Wih  = (const float*)d_in[5];
  const float* Whh  = (const float*)d_in[6];
  const float* bih  = (const float*)d_in[7];
  const float* bhh  = (const float*)d_in[8];
  const float* Wout = (const float*)d_in[9];
  const float* bout = (const float*)d_in[10];
  float* out = (float*)d_out;

  char* ws = (char*)d_ws;
  unsigned short* Wcat  = (unsigned short*)(ws + OFF_WCAT);
  unsigned short* WoutB = (unsigned short*)(ws + OFF_WOUT);
  unsigned short* EmbB  = (unsigned short*)(ws + OFF_EMB);
  float*          bias  = (float*)(ws + OFF_BIAS);
  unsigned short* xh0   = (unsigned short*)(ws + OFF_XH0);
  unsigned short* xh1   = (unsigned short*)(ws + OFF_XH1);
  float*          Cst   = (float*)(ws + OFF_C);
  unsigned short* xh[2] = { xh0, xh1 };

  k_prep_wcat<<<FOURH * K2 / 256, 256, 0, stream>>>(Wih, Whh, Wcat);
  k_prep_misc<<<(2 * V * H + FOURH + 255) / 256, 256, 0, stream>>>(
      Wout, Emb, bih, bhh, WoutB, EmbB, bias);
  k_init_state<<<B * H / 256, 256, 0, stream>>>(encH, encC, Emb, xh0, Cst);

  for (int t = 0; t < T; ++t) {
    unsigned short* in  = xh[t & 1];
    unsigned short* nxt = xh[(t + 1) & 1];
    k_lstm_step<<<dim3(B / 64, H / 32), 256, 0, stream>>>(in, nxt, Wcat, bias, Cst);
    k_logits_argmax<<<B / 16, 256, 0, stream>>>(nxt, WoutB, bout, EmbB,
                                                out + (size_t)t * B * V);
  }
}
